// MyNet_Torch_29094108463398
// MI455X (gfx1250) — compile-verified, matrix operand order checked by reading, not run on hardware
//
#include <hip/hip_runtime.h>

typedef __attribute__((ext_vector_type(16))) _Float16 v16h;
typedef __attribute__((ext_vector_type(8)))  _Float16 h8v;
typedef __attribute__((ext_vector_type(8)))  float    v8f;

#define CN 100000
#define CR 1000
#define C2R 2000
#define CD 128
#define CE 800000
#define CDD (128 * 128)

// ---------------------------------------------------------------------------
// Utility kernels
// ---------------------------------------------------------------------------
__global__ void zero_kernel(float* __restrict__ p, long long n) {
  long long i = (long long)blockIdx.x * blockDim.x + threadIdx.x;
  long long stride = (long long)gridDim.x * blockDim.x;
  for (; i < n; i += stride) p[i] = 0.0f;
}

__global__ void copy_kernel(const float* __restrict__ s, float* __restrict__ d, long long n) {
  long long i = (long long)blockIdx.x * blockDim.x + threadIdx.x;
  long long stride = (long long)gridDim.x * blockDim.x;
  for (; i < n; i += stride) d[i] = s[i];
}

// Column sums / sums-of-squares over [rows,128]; blockDim.x==128, each block
// reduces 256 rows locally then does one float atomic per column.
__global__ void colstats_kernel(const float* __restrict__ x, int rows,
                                float* __restrict__ sum, float* __restrict__ sumsq) {
  int col = threadIdx.x;
  long long r0 = (long long)blockIdx.x * 256;
  long long rem = (long long)rows - r0;
  int lim = rem < 256 ? (int)rem : 256;
  float s = 0.0f, q = 0.0f;
  for (int r = 0; r < lim; ++r) {
    float v = x[(r0 + r) * CD + col];
    s += v; q += v * v;
  }
  unsafeAtomicAdd(&sum[col], s);
  unsafeAtomicAdd(&sumsq[col], q);
}

// BN(x) = x*scale + shift with biased batch stats (matches jnp.var ddof=0).
__global__ void bn_prep_kernel(const float* __restrict__ sum, const float* __restrict__ sumsq,
                               float invN, const float* __restrict__ gamma,
                               const float* __restrict__ beta,
                               float* __restrict__ scale, float* __restrict__ shift) {
  int k = threadIdx.x;
  float m = sum[k] * invN;
  float var = sumsq[k] * invN - m * m;
  float s = gamma[k] * rsqrtf(var + 1e-5f);
  scale[k] = s;
  shift[k] = beta[k] - m * s;
}

// ---------------------------------------------------------------------------
// WMMA GEMM: C[M,128] = act( BN(A[M,128]) @ W[128,128] + bias )
// BN applied iff scale != nullptr. act: 0 = identity, 1 = sigmoid.
// Block = 256 threads = 8 waves; each wave owns a 16-row output strip.
// W is staged to LDS transposed (col-major, padded) so B fragments are two
// ds_load_b128 per tile per lane.
// ---------------------------------------------------------------------------
#define WT_STRIDE 136  // 128 + 8 halves padding

__launch_bounds__(256)
__global__ void gemm128_wmma(const float* __restrict__ A, int M,
                             const float* __restrict__ W,
                             const float* __restrict__ bias,
                             const float* __restrict__ scale,
                             const float* __restrict__ shift,
                             float* __restrict__ C, int act) {
  __shared__ __align__(16) _Float16 Wt[CD * WT_STRIDE];
  // Stage W[k][n] -> Wt[n][k] as f16 (coalesced global reads).
  for (int idx = threadIdx.x; idx < CDD; idx += 256) {
    int k = idx >> 7, n = idx & 127;
    Wt[n * WT_STRIDE + k] = (_Float16)W[idx];
  }
  __syncthreads();

  const int lane   = threadIdx.x & 31;
  const int wave   = threadIdx.x >> 5;
  const int laneHi = lane >> 4;      // 0: lanes 0-15, 1: lanes 16-31
  const int lrow   = lane & 15;

  long long baseRow = ((long long)blockIdx.x * 8 + wave) * 16;
  if (baseRow >= M) return;  // wave-uniform; EXEC stays all-ones for WMMA

  long long arow = baseRow + lrow;
  if (arow > M - 1) arow = M - 1;    // clamp loads; stores guarded below
  const float* aBase = A + arow * CD;

  v8f acc[8];
#pragma unroll
  for (int nt = 0; nt < 8; ++nt) {
    float bv = bias ? bias[nt * 16 + lrow] : 0.0f;  // per-column bias (C: N = lane&15)
#pragma unroll
    for (int j = 0; j < 8; ++j) acc[nt][j] = bv;
  }

#pragma unroll
  for (int kb = 0; kb < 4; ++kb) {
    // A fragment per ISA layout: halves 0-7 -> K = kbase+0..7,
    // halves 8-15 -> K = kbase+16..23, kbase = kb*32 + (laneHi?8:0).
    int kbase = kb * 32 + (laneHi ? 8 : 0);
    const float4* ap = reinterpret_cast<const float4*>(aBase + kbase);
    float4 p0 = ap[0];
    float4 p1 = ap[1];
    float4 p2 = ap[4];
    float4 p3 = ap[5];
    float av[16] = {p0.x, p0.y, p0.z, p0.w, p1.x, p1.y, p1.z, p1.w,
                    p2.x, p2.y, p2.z, p2.w, p3.x, p3.y, p3.z, p3.w};
    if (scale) {
#pragma unroll
      for (int i = 0; i < 16; ++i) {
        int k = kbase + (i < 8 ? i : 8 + i);
        av[i] = av[i] * scale[k] + shift[k];
      }
    }
    v16h afrag;
#pragma unroll
    for (int i = 0; i < 16; ++i) afrag[i] = (_Float16)av[i];

#pragma unroll
    for (int nt = 0; nt < 8; ++nt) {
      int col = nt * 16 + lrow;                 // B fragment: lane holds column N
      int b2 = col * WT_STRIDE + kbase;
      h8v blo = *reinterpret_cast<const h8v*>(&Wt[b2]);
      h8v bhi = *reinterpret_cast<const h8v*>(&Wt[b2 + 16]);
      v16h bfrag;
#pragma unroll
      for (int i = 0; i < 8; ++i) { bfrag[i] = blo[i]; bfrag[8 + i] = bhi[i]; }
      acc[nt] = __builtin_amdgcn_wmma_f32_16x16x32_f16(
          false, afrag, false, bfrag, (short)0, acc[nt], false, false);
    }
  }

  // Epilogue: C/D layout — element j: M = baseRow + j + (laneHi?8:0), N = nt*16 + lrow.
#pragma unroll
  for (int nt = 0; nt < 8; ++nt) {
    int col = nt * 16 + lrow;
#pragma unroll
    for (int j = 0; j < 8; ++j) {
      long long m = baseRow + j + (laneHi ? 8 : 0);
      if (m < M) {
        float v = acc[nt][j];
        if (act == 1) v = 1.0f / (1.0f + __expf(-v));
        C[m * CD + col] = v;
      }
    }
  }
}

// ---------------------------------------------------------------------------
// Edge scatter: h[dst] += gate_table[edge_attr] * x[src]. One wave per edge,
// each lane handles 4 consecutive dims (float4 gathers, 4 fp32 atomics).
// ---------------------------------------------------------------------------
__launch_bounds__(256)
__global__ void scatter_rgc_kernel(const float* __restrict__ x, const float* __restrict__ gtab,
                                   const int* __restrict__ srcI, const int* __restrict__ dstI,
                                   const int* __restrict__ eattr, int E,
                                   float* __restrict__ h) {
  long long gid = (long long)blockIdx.x * blockDim.x + threadIdx.x;
  long long e = gid >> 5;
  int lane = (int)(gid & 31);
  if (e >= E) return;
  int s = srcI[e], d = dstI[e], r = eattr[e];
  int base = lane * 4;
  float4 g  = *reinterpret_cast<const float4*>(gtab + (long long)r * CD + base);
  float4 xv = *reinterpret_cast<const float4*>(x + (long long)s * CD + base);
  float* hp = h + (long long)d * CD + base;
  unsafeAtomicAdd(hp + 0, g.x * xv.x);
  unsafeAtomicAdd(hp + 1, g.y * xv.y);
  unsafeAtomicAdd(hp + 2, g.z * xv.z);
  unsafeAtomicAdd(hp + 3, g.w * xv.w);
}

// Normalize each 128-wide row of src, write into dst[row*stride + off + t].
__global__ void normalize_copy_kernel(const float* __restrict__ src, float* __restrict__ dst,
                                      int dstStride, int colOff) {
  __shared__ float red[CD];
  long long row = blockIdx.x;
  int t = threadIdx.x;
  float v = src[row * CD + t];
  red[t] = v * v;
  __syncthreads();
  for (int s = 64; s > 0; s >>= 1) {
    if (t < s) red[t] += red[t + s];
    __syncthreads();
  }
  float dnm = fmaxf(sqrtf(red[0]), 1e-12f);
  dst[row * (long long)dstStride + colOff + t] = v / dnm;
}

// In-place row L2 renormalization for rows of `width` floats.
__global__ void renorm_rows_kernel(float* __restrict__ data, int width) {
  __shared__ float red[256];
  long long row = blockIdx.x;
  int t = threadIdx.x;
  float q = 0.0f;
  for (int c = t; c < width; c += 256) {
    float v = data[row * (long long)width + c];
    q += v * v;
  }
  red[t] = q;
  __syncthreads();
  for (int s = 128; s > 0; s >>= 1) {
    if (t < s) red[t] += red[t + s];
    __syncthreads();
  }
  float inv = 1.0f / fmaxf(sqrtf(red[0]), 1e-12f);
  for (int c = t; c < width; c += 256) data[row * (long long)width + c] *= inv;
}

// ---------------------------------------------------------------------------
extern "C" void kernel_launch(void* const* d_in, const int* in_sizes, int n_in,
                              void* d_out, int out_size, void* d_ws, size_t ws_size,
                              hipStream_t stream) {
  const float* ent_emb  = (const float*)d_in[0];
  const float* rel_emb  = (const float*)d_in[1];
  const float* W_inv    = (const float*)d_in[2];
  const float* b_inv    = (const float*)d_in[3];
  const float* bn_gamma = (const float*)d_in[4];
  const float* bn_beta  = (const float*)d_in[5];
  const float* rel_W    = (const float*)d_in[6];
  const float* rel_b    = (const float*)d_in[7];
  const float* ent_W    = (const float*)d_in[8];
  const float* ent_b    = (const float*)d_in[9];
  const float* s_W      = (const float*)d_in[10];
  const float* s_b      = (const float*)d_in[11];
  const float* conv_Wg  = (const float*)d_in[12];
  const float* conv_bg  = (const float*)d_in[13];
  const float* sconv_Wg = (const float*)d_in[14];
  const float* sconv_bg = (const float*)d_in[15];
  const int* edge_index = (const int*)d_in[16];
  const int* cross_edge = (const int*)d_in[17];
  const int* edge_attr  = (const int*)d_in[18];

  const long long ND = (long long)CN * CD;
  const long long RD2 = (long long)C2R * CD;

  float* ws = (float*)d_ws;
  float* relAll = ws;              // [2R,128]
  float* g1     = relAll + RD2;    // gate table conv   [2R,128]
  float* g2     = g1 + RD2;        // gate table sconv  [2R,128]
  float* bufA   = g2 + RD2;        // out stream        [N,128]
  float* bufB   = bufA + ND;       // s_out stream      [N,128]
  float* bufC   = bufB + ND;       // x1                [N,128]
  float* bufD   = bufC + ND;       // x2                [N,128]
  float* stats  = bufD + ND;       // sumA,sqA,sumB,sqB,scaleA,shiftA,scaleB,shiftB (8*128)

  float* final_embed = (float*)d_out;                         // [N,640]
  float* rel_final   = final_embed + (long long)CN * 640;     // [2R,384]
  float* last_ent    = rel_final + (long long)C2R * 384;      // [N,256]
  float* last_rel    = last_ent + (long long)CN * 256;        // [2R,128]

  const int gemmBlocksN   = (CN + 127) / 128;   // 782
  const int gemmBlocksRel = (C2R + 127) / 128;  // 16
  const int gemmBlocksR   = (CR + 127) / 128;   // 8
  const int statBlocks    = (CN + 255) / 256;   // 391
  const int scatBlocks    = (int)(((long long)CE * 32 + 255) / 256);

  // ---- rel_all = [rel_emb ; rel_emb @ W_inv + b_inv] ----
  copy_kernel<<<512, 256, 0, stream>>>(rel_emb, relAll, (long long)CR * CD);
  gemm128_wmma<<<gemmBlocksR, 256, 0, stream>>>(rel_emb, CR, W_inv, b_inv,
                                                nullptr, nullptr, relAll + (long long)CR * CD, 0);
  // rel_list entry 0
  normalize_copy_kernel<<<C2R, CD, 0, stream>>>(relAll, rel_final, 384, 0);
  // out_list entry 0
  normalize_copy_kernel<<<CN, CD, 0, stream>>>(ent_emb, final_embed, 640, 0);

  for (int i = 0; i < 2; ++i) {
    const float* outp  = (i == 0) ? ent_emb : bufA;
    const float* soutp = (i == 0) ? ent_emb : bufB;
    const float* gam = bn_gamma + i * CD;
    const float* bet = bn_beta + i * CD;

    // ---- BatchNorm stats (fused into GEMM A-load as per-column affine) ----
    zero_kernel<<<2, 256, 0, stream>>>(stats, 4 * CD);
    colstats_kernel<<<statBlocks, CD, 0, stream>>>(outp, CN, stats, stats + CD);
    colstats_kernel<<<statBlocks, CD, 0, stream>>>(soutp, CN, stats + 2 * CD, stats + 3 * CD);
    bn_prep_kernel<<<1, CD, 0, stream>>>(stats, stats + CD, 1.0f / CN, gam, bet,
                                         stats + 4 * CD, stats + 5 * CD);
    bn_prep_kernel<<<1, CD, 0, stream>>>(stats + 2 * CD, stats + 3 * CD, 1.0f / CN, gam, bet,
                                         stats + 6 * CD, stats + 7 * CD);

    // ---- relation update (in place; each wave reads only its own row strip) ----
    gemm128_wmma<<<gemmBlocksRel, 256, 0, stream>>>(relAll, C2R, rel_W + i * CDD,
                                                    rel_b + i * CD, nullptr, nullptr, relAll, 0);
    // ---- gate tables: sigmoid(rel_all @ Wg + bg) (gather hoisted out of E) ----
    gemm128_wmma<<<gemmBlocksRel, 256, 0, stream>>>(relAll, C2R, conv_Wg + i * CDD,
                                                    conv_bg + i * CD, nullptr, nullptr, g1, 1);
    gemm128_wmma<<<gemmBlocksRel, 256, 0, stream>>>(relAll, C2R, sconv_Wg + i * CDD,
                                                    sconv_bg + i * CD, nullptr, nullptr, g2, 1);

    // ---- entity GEMMs with fused BN: x1 = BN(out)@entW+b, x2 = BN(s_out)@sW+b ----
    gemm128_wmma<<<gemmBlocksN, 256, 0, stream>>>(outp, CN, ent_W + i * CDD, ent_b + i * CD,
                                                  stats + 4 * CD, stats + 5 * CD, bufC, 0);
    gemm128_wmma<<<gemmBlocksN, 256, 0, stream>>>(soutp, CN, s_W + i * CDD, s_b + i * CD,
                                                  stats + 6 * CD, stats + 7 * CD, bufD, 0);

    // ---- gated scatter-sum; streams swap: new out = h2, new s_out = h1 ----
    zero_kernel<<<4096, 256, 0, stream>>>(bufA, ND);
    zero_kernel<<<4096, 256, 0, stream>>>(bufB, ND);
    scatter_rgc_kernel<<<scatBlocks, 256, 0, stream>>>(bufC, g1, edge_index, edge_index + CE,
                                                       edge_attr, CE, bufB);  // h1
    scatter_rgc_kernel<<<scatBlocks, 256, 0, stream>>>(bufD, g2, cross_edge, cross_edge + CE,
                                                       edge_attr, CE, bufA);  // h2

    // ---- normalized list entries straight into output column slots ----
    int e1 = 1 + 2 * i, e2 = 2 + 2 * i;
    normalize_copy_kernel<<<CN, CD, 0, stream>>>(bufA, final_embed, 640, e1 * CD);
    normalize_copy_kernel<<<CN, CD, 0, stream>>>(bufB, final_embed, 640, e2 * CD);
    normalize_copy_kernel<<<C2R, CD, 0, stream>>>(relAll, rel_final, 384, (i + 1) * CD);
    if (i == 1) {
      normalize_copy_kernel<<<CN, CD, 0, stream>>>(bufA, last_ent, 256, 0);
      normalize_copy_kernel<<<CN, CD, 0, stream>>>(bufB, last_ent, 256, CD);
      normalize_copy_kernel<<<C2R, CD, 0, stream>>>(relAll, last_rel, CD, 0);
    }
  }

  // ---- final concat renormalizations ----
  renorm_rows_kernel<<<CN, 256, 0, stream>>>(final_embed, 640);
  renorm_rows_kernel<<<C2R, 256, 0, stream>>>(rel_final, 384);
  renorm_rows_kernel<<<CN, 256, 0, stream>>>(last_ent, 256);
  // last_rel is rel_list[-1]: already a per-128 normalized copy, no renorm.
}
